// Head_29377576304777
// MI455X (gfx1250) — matrix operand order repaired and checked by reading, not compiled, not run on hardware
//
#include <hip/hip_runtime.h>

// Problem dims (fixed by the reference)
constexpr int BB = 32;    // batch
constexpr int TT = 2048;  // tokens
constexpr int CC = 1024;  // embed
constexpr int HH = 128;   // head size

typedef __bf16 bf16_t;
typedef __attribute__((ext_vector_type(16))) bf16_t v16bf;
typedef __attribute__((ext_vector_type(8)))  bf16_t v8bf;
typedef __attribute__((ext_vector_type(8)))  float  v8f;

static __device__ __forceinline__ v16bf pack16(v8bf lo, v8bf hi) {
  v16bf r;
#pragma unroll
  for (int i = 0; i < 8; i++) { r[i] = lo[i]; r[i + 8] = hi[i]; }
  return r;
}

static __device__ __forceinline__ v16bf cvt16(float4 f0, float4 f1,
                                              float4 f2, float4 f3) {
  v16bf a;
  a[0] = (bf16_t)f0.x;  a[1] = (bf16_t)f0.y;  a[2] = (bf16_t)f0.z;  a[3] = (bf16_t)f0.w;
  a[4] = (bf16_t)f1.x;  a[5] = (bf16_t)f1.y;  a[6] = (bf16_t)f1.z;  a[7] = (bf16_t)f1.w;
  a[8] = (bf16_t)f2.x;  a[9] = (bf16_t)f2.y;  a[10] = (bf16_t)f2.z; a[11] = (bf16_t)f2.w;
  a[12] = (bf16_t)f3.x; a[13] = (bf16_t)f3.y; a[14] = (bf16_t)f3.z; a[15] = (bf16_t)f3.w;
  return a;
}

static __device__ __forceinline__ v8f wmma_bf16(v16bf a, v16bf b, v8f c) {
  // D = A(16x32 bf16) * B(32x16 bf16) + C(16x16 f32)
  return __builtin_amdgcn_wmma_f32_16x16x32_bf16(
      false, a, false, b, (short)0, c, false, false);
}

// ---------------------------------------------------------------------------
// Kernel 1: q/k/v = x @ W{q,k,v}, bf16 output into workspace.
// GEMM M=B*T=65536, K=C=1024, N=H=128. 256 threads = 8 waves, 16 rows/wave.
// Double-buffered LDS W^T tile; next X/W loads issued before the WMMA block,
// their waits land after it (latency hidden under compute).
// ---------------------------------------------------------------------------
__global__ __launch_bounds__(256)
void qkv_proj(const float* __restrict__ x,
              const float* __restrict__ Wq,
              const float* __restrict__ Wk,
              const float* __restrict__ Wv,
              bf16_t* __restrict__ qkv) {
  __shared__ bf16_t Wlds[2][HH][32];  // W tile transposed: [col][k], 2 x 8 KB

  const int proj = blockIdx.y;
  const float* W = (proj == 0) ? Wq : (proj == 1) ? Wk : Wv;
  bf16_t* out = qkv + (size_t)proj * (size_t)BB * TT * HH;

  const int tid    = threadIdx.x;
  const int wave   = tid >> 5;
  const int lane   = tid & 31;
  const int laneN  = lane & 15;
  const int laneHi = lane >> 4;
  const int m0     = blockIdx.x * 128 + wave * 16;
  const int sk     = tid >> 3;         // staging: k row 0..31
  const int sn0    = (tid & 7) * 16;   // staging: col base

  v8f acc[8];
#pragma unroll
  for (int n = 0; n < 8; n++) acc[n] = v8f{};

  // A-fragment base: lane<16 -> K {0-7,16-23}; lane>=16 -> {8-15,24-31}
  const float* xrow = x + (size_t)(m0 + laneN) * CC + laneHi * 8;

  // ---- prologue: stage W tile 0, load X chunk 0 ----
  {
    const float* wp = W + (size_t)sk * HH + sn0;
#pragma unroll
    for (int i = 0; i < 16; i++) Wlds[0][sn0 + i][sk] = (bf16_t)wp[i];
  }
  float4 x0 = *(const float4*)(xrow);
  float4 x1 = *(const float4*)(xrow + 4);
  float4 x2 = *(const float4*)(xrow + 16);
  float4 x3 = *(const float4*)(xrow + 20);
  __syncthreads();

  for (int k0 = 0; k0 < CC; k0 += 32) {
    const int  buf     = (k0 >> 5) & 1;
    const bool doStage = (k0 + 32) < CC;

    // Issue next-tile global loads now; waits happen at the uses below.
    float4 w0, w1, w2, w3, nx0, nx1, nx2, nx3;
    if (doStage) {
      const float* wp = W + (size_t)(k0 + 32 + sk) * HH + sn0;
      w0 = *(const float4*)(wp);      w1 = *(const float4*)(wp + 4);
      w2 = *(const float4*)(wp + 8);  w3 = *(const float4*)(wp + 12);
      const float* xp = xrow + k0 + 32;
      nx0 = *(const float4*)(xp);      nx1 = *(const float4*)(xp + 4);
      nx2 = *(const float4*)(xp + 16); nx3 = *(const float4*)(xp + 20);
      __builtin_prefetch(xp + 32, 0, 1);  // one more K-step ahead
    }

    // A fragment from X loaded last iteration (latency already hidden).
    const v16bf a = cvt16(x0, x1, x2, x3);

    // Preload all 8 B fragments into distinct registers, then WMMA chain.
    v16bf bfrag[8];
#pragma unroll
    for (int n = 0; n < 8; n++) {
      const bf16_t* bp = &Wlds[buf][n * 16 + laneN][laneHi * 8];
      bfrag[n] = pack16(*(const v8bf*)bp, *(const v8bf*)(bp + 16));
    }
#pragma unroll
    for (int n = 0; n < 8; n++) acc[n] = wmma_bf16(a, bfrag[n], acc[n]);

    if (doStage) {
      // W stores (wait for W loads lands here, after the WMMAs).
      Wlds[buf ^ 1][sn0 +  0][sk] = (bf16_t)w0.x;
      Wlds[buf ^ 1][sn0 +  1][sk] = (bf16_t)w0.y;
      Wlds[buf ^ 1][sn0 +  2][sk] = (bf16_t)w0.z;
      Wlds[buf ^ 1][sn0 +  3][sk] = (bf16_t)w0.w;
      Wlds[buf ^ 1][sn0 +  4][sk] = (bf16_t)w1.x;
      Wlds[buf ^ 1][sn0 +  5][sk] = (bf16_t)w1.y;
      Wlds[buf ^ 1][sn0 +  6][sk] = (bf16_t)w1.z;
      Wlds[buf ^ 1][sn0 +  7][sk] = (bf16_t)w1.w;
      Wlds[buf ^ 1][sn0 +  8][sk] = (bf16_t)w2.x;
      Wlds[buf ^ 1][sn0 +  9][sk] = (bf16_t)w2.y;
      Wlds[buf ^ 1][sn0 + 10][sk] = (bf16_t)w2.z;
      Wlds[buf ^ 1][sn0 + 11][sk] = (bf16_t)w2.w;
      Wlds[buf ^ 1][sn0 + 12][sk] = (bf16_t)w3.x;
      Wlds[buf ^ 1][sn0 + 13][sk] = (bf16_t)w3.y;
      Wlds[buf ^ 1][sn0 + 14][sk] = (bf16_t)w3.z;
      Wlds[buf ^ 1][sn0 + 15][sk] = (bf16_t)w3.w;
      x0 = nx0; x1 = nx1; x2 = nx2; x3 = nx3;
    }
    __syncthreads();
  }

  const float scale = (proj == 0) ? 0.03125f : 1.0f;  // C^-0.5 folded into q
#pragma unroll
  for (int r = 0; r < 8; r++) {
    const int row = m0 + r + laneHi * 8;
#pragma unroll
    for (int n = 0; n < 8; n++)
      out[(size_t)row * HH + n * 16 + laneN] = (bf16_t)(acc[n][r] * scale);
  }
}

// ---------------------------------------------------------------------------
// Kernel 2: flash attention. grid = (T/128, B), 256 threads (8 waves).
// K and V^T tiles double-buffered in LDS (coalesced cooperative staging,
// shared by all 8 waves); stage loads issued before compute, stored after.
// ---------------------------------------------------------------------------
__global__ __launch_bounds__(256)
void attn_fwd(const bf16_t* __restrict__ qkv, float* __restrict__ out) {
  __shared__ bf16_t klds[2][32][HH];    // K tile row-major [key][h], 2 x 8 KB
  __shared__ bf16_t vtl[2][HH][32];     // V^T tile [h][key],         2 x 8 KB
  __shared__ bf16_t plds[8][16][32];    // per-wave P scratch,            8 KB

  const int b      = blockIdx.y;
  const int qblk   = blockIdx.x;
  const int tid    = threadIdx.x;
  const int wave   = tid >> 5;
  const int lane   = tid & 31;
  const int laneN  = lane & 15;
  const int laneHi = lane >> 4;
  const int q0     = qblk * 128 + wave * 16;

  const size_t plane = (size_t)BB * TT * HH;
  const bf16_t* qb = qkv + (size_t)b * TT * HH;              // q (pre-scaled)
  const bf16_t* kp = qkv + plane + (size_t)b * TT * HH;      // k
  const bf16_t* vp = qkv + 2 * plane + (size_t)b * TT * HH;  // v

  const int skey = tid >> 3;          // staging: key 0..31
  const int sh0  = (tid & 7) * 16;    // staging: h base
  const bf16_t* kstage = kp + (size_t)skey * HH + sh0;
  const bf16_t* vstage = vp + (size_t)skey * HH + sh0;

  // Q A-fragments for the wave's 16 rows: 4 chunks of K=32 over H=128.
  v16bf qa[4];
  {
    const bf16_t* qrow = qb + (size_t)(q0 + laneN) * HH;
#pragma unroll
    for (int c = 0; c < 4; c++) {
      const bf16_t* p = qrow + c * 32 + laneHi * 8;
      qa[c] = pack16(*(const v8bf*)p, *(const v8bf*)(p + 16));
    }
  }

  v8f acc[8];
#pragma unroll
  for (int n = 0; n < 8; n++) acc[n] = v8f{};
  float mrow[8], lrow[8];
#pragma unroll
  for (int r = 0; r < 8; r++) { mrow[r] = -__builtin_inff(); lrow[r] = 0.0f; }

  const int nkb = (qblk + 1) * 4;  // 32-key blocks up to the causal limit

  // ---- prologue: stage key block 0 into buffer 0 ----
  {
    const v8bf ka = *(const v8bf*)(kstage);
    const v8bf kb = *(const v8bf*)(kstage + 8);
    const v8bf va = *(const v8bf*)(vstage);
    const v8bf vb = *(const v8bf*)(vstage + 8);
    *(v8bf*)&klds[0][skey][sh0]     = ka;
    *(v8bf*)&klds[0][skey][sh0 + 8] = kb;
#pragma unroll
    for (int i = 0; i < 8; i++) {
      vtl[0][sh0 + i][skey]     = va[i];
      vtl[0][sh0 + 8 + i][skey] = vb[i];
    }
  }
  __syncthreads();

  for (int kblk = 0; kblk < nkb; kblk++) {
    const int  kbase   = kblk * 32;
    const int  buf     = kblk & 1;
    const bool doStage = (kblk + 1) < nkb;

    // Issue next-block global loads now; waits land after compute.
    v8bf ka, kb, va, vb;
    if (doStage) {
      const bf16_t* ks = kstage + (size_t)(kbase + 32) * HH;
      const bf16_t* vs = vstage + (size_t)(kbase + 32) * HH;
      ka = *(const v8bf*)(ks);  kb = *(const v8bf*)(ks + 8);
      va = *(const v8bf*)(vs);  vb = *(const v8bf*)(vs + 8);
      if (kblk + 2 < nkb) {                       // -> global_prefetch_b8
        __builtin_prefetch(ks + (size_t)32 * HH, 0, 1);
        __builtin_prefetch(vs + (size_t)32 * HH, 0, 1);
      }
    }

    if (kbase <= q0 + 15) {  // wave-uniform: EXEC all-ones around WMMA
      // ---- S = Q . K^T : preload all K fragments, 2 interleaved chains ----
      v16bf kf[2][4];
#pragma unroll
      for (int t = 0; t < 2; t++)
#pragma unroll
        for (int c = 0; c < 4; c++) {
          const bf16_t* p = &klds[buf][t * 16 + laneN][c * 32 + laneHi * 8];
          kf[t][c] = pack16(*(const v8bf*)p, *(const v8bf*)(p + 16));
        }
      v8f s[2];
      s[0] = v8f{}; s[1] = v8f{};
#pragma unroll
      for (int c = 0; c < 4; c++) {
        s[0] = wmma_bf16(qa[c], kf[0][c], s[0]);
        s[1] = wmma_bf16(qa[c], kf[1][c], s[1]);
      }

      // Preload V fragments early; ds latency hides under softmax VALU.
      v16bf vf[8];
#pragma unroll
      for (int n = 0; n < 8; n++) {
        const bf16_t* p = &vtl[buf][n * 16 + laneN][laneHi * 8];
        vf[n] = pack16(*(const v8bf*)p, *(const v8bf*)(p + 16));
      }

      // ---- causal mask (diagonal super-blocks only) ----
      if (kbase + 31 > q0) {
#pragma unroll
        for (int t = 0; t < 2; t++) {
          const int key = kbase + t * 16 + laneN;
#pragma unroll
          for (int r = 0; r < 8; r++) {
            const int qq = q0 + r + laneHi * 8;
            if (key > qq) s[t][r] = -__builtin_inff();
          }
        }
      }

      // ---- online softmax (rows live in 16-lane halves) ----
#pragma unroll
      for (int r = 0; r < 8; r++) {
        float mx = fmaxf(s[0][r], s[1][r]);
#pragma unroll
        for (int off = 8; off >= 1; off >>= 1)
          mx = fmaxf(mx, __shfl_xor(mx, off, 32));
        const float mnew = fmaxf(mrow[r], mx);
        const float corr = __expf(mrow[r] - mnew);
        const float p0 = __expf(s[0][r] - mnew);
        const float p1 = __expf(s[1][r] - mnew);
        s[0][r] = p0; s[1][r] = p1;
        float rs = p0 + p1;
#pragma unroll
        for (int off = 8; off >= 1; off >>= 1)
          rs += __shfl_xor(rs, off, 32);
        lrow[r] = lrow[r] * corr + rs;
        mrow[r] = mnew;
#pragma unroll
        for (int n = 0; n < 8; n++) acc[n][r] *= corr;
      }

      // ---- P: D-layout -> per-wave LDS -> A-layout (bf16) ----
#pragma unroll
      for (int t = 0; t < 2; t++)
#pragma unroll
        for (int r = 0; r < 8; r++)
          plds[wave][r + laneHi * 8][t * 16 + laneN] = (bf16_t)s[t][r];

      v16bf pa;
      {
        const bf16_t* pp = &plds[wave][laneN][laneHi * 8];
        pa = pack16(*(const v8bf*)pp, *(const v8bf*)(pp + 16));
      }

      // ---- O += P . V : 8 independent WMMAs ----
#pragma unroll
      for (int n = 0; n < 8; n++)
        acc[n] = wmma_bf16(pa, vf[n], acc[n]);
    }

    if (doStage) {
      // Stores wait for the stage loads here, after the compute block.
      *(v8bf*)&klds[buf ^ 1][skey][sh0]     = ka;
      *(v8bf*)&klds[buf ^ 1][skey][sh0 + 8] = kb;
#pragma unroll
      for (int i = 0; i < 8; i++) {
        vtl[buf ^ 1][sh0 + i][skey]     = va[i];
        vtl[buf ^ 1][sh0 + 8 + i][skey] = vb[i];
      }
    }
    __syncthreads();
  }

  // ---- epilogue: O /= l, store f32 ----
#pragma unroll
  for (int r = 0; r < 8; r++) {
    const float inv = 1.0f / lrow[r];
    const size_t row = (size_t)b * TT + q0 + r + laneHi * 8;
#pragma unroll
    for (int n = 0; n < 8; n++)
      out[row * HH + n * 16 + laneN] = acc[n][r] * inv;
  }
}

extern "C" void kernel_launch(void* const* d_in, const int* in_sizes, int n_in,
                              void* d_out, int out_size, void* d_ws, size_t ws_size,
                              hipStream_t stream) {
  (void)in_sizes; (void)n_in; (void)out_size; (void)ws_size;
  const float* x  = (const float*)d_in[0];
  const float* Wq = (const float*)d_in[1];
  const float* Wk = (const float*)d_in[2];
  const float* Wv = (const float*)d_in[3];
  float* out = (float*)d_out;
  bf16_t* qkv = (bf16_t*)d_ws;  // 3 * B*T*H bf16 = 48 MB scratch

  dim3 gProj(BB * TT / 128, 3);
  qkv_proj<<<gProj, 256, 0, stream>>>(x, Wq, Wk, Wv, qkv);

  dim3 gAttn(TT / 128, BB);
  attn_fwd<<<gAttn, 256, 0, stream>>>(qkv, out);
}
